// GCN_66348654789162
// MI455X (gfx1250) — matrix operand order repaired and checked by reading, not compiled, not run on hardware
//
#include <hip/hip_runtime.h>
#include <hip/hip_bf16.h>

#define NNODES  100000
#define NEDGES  1600000
#define NGRAPHS 2048
#define DIMH    128
#define DIMFC   256
#define DIMOUT  8
#define KTOT    256        // concat K: [agg | h]
#define WSTRIDE 264        // padded LDS k-stride (halves), 16B aligned, bank-friendly

typedef __attribute__((ext_vector_type(16))) _Float16 v16h;
typedef __attribute__((ext_vector_type(8)))  _Float16 v8h;
typedef __attribute__((ext_vector_type(4)))  _Float16 v4h;
typedef __attribute__((ext_vector_type(8)))  float    v8f;

// ---------------- utility ----------------
__global__ __launch_bounds__(256) void k_zero_i(int* p, int n) {
  int i = blockIdx.x * 256 + threadIdx.x;
  if (i < n) p[i] = 0;
}
__global__ __launch_bounds__(256) void k_zero_f(float* p, int n) {
  int i = blockIdx.x * 256 + threadIdx.x;
  if (i < n) p[i] = 0.0f;
}

// ---------------- CSR build (edge_index is layer-invariant) ----------------
__global__ __launch_bounds__(256) void k_hist(const int* __restrict__ dst, int* __restrict__ cnt) {
  int e = blockIdx.x * 256 + threadIdx.x;
  if (e < NEDGES) atomicAdd(&cnt[dst[e]], 1);
}

__global__ __launch_bounds__(512) void k_scan1(const int* __restrict__ cnt, int* __restrict__ excl,
                                               int* __restrict__ bsums, int n) {
  __shared__ int tmp[512];
  int t = threadIdx.x;
  int g = blockIdx.x * 512 + t;
  int v = (g < n) ? cnt[g] : 0;
  tmp[t] = v;
  __syncthreads();
  for (int off = 1; off < 512; off <<= 1) {
    int y = (t >= off) ? tmp[t - off] : 0;
    __syncthreads();
    tmp[t] += y;
    __syncthreads();
  }
  if (g < n) excl[g] = tmp[t] - v;          // exclusive within block
  if (t == 511) bsums[blockIdx.x] = tmp[511];
}

__global__ void k_scan2(int* bsums, int nb) {
  if (threadIdx.x == 0 && blockIdx.x == 0) {
    int run = 0;
    for (int i = 0; i < nb; ++i) { int x = bsums[i]; bsums[i] = run; run += x; }
  }
}

__global__ __launch_bounds__(256) void k_scan3(int* __restrict__ rowptr, int* __restrict__ cursor,
                                               const int* __restrict__ bsums, int n, int total) {
  int g = blockIdx.x * 256 + threadIdx.x;
  if (g < n) {
    int r = rowptr[g] + bsums[g >> 9];
    rowptr[g] = r;
    cursor[g] = r;
  } else if (g == n) {
    rowptr[n] = total;
  }
}

__global__ __launch_bounds__(256) void k_fill(const int* __restrict__ src, const int* __restrict__ dst,
                                              int* __restrict__ cursor, int* __restrict__ eidx) {
  int e = blockIdx.x * 256 + threadIdx.x;
  if (e < NEDGES) {
    int pos = atomicAdd(&cursor[dst[e]], 1);
    eidx[pos] = src[e];
  }
}

// ---------------- f32 -> f16 activation convert ----------------
__global__ __launch_bounds__(256) void k_cvt_x(const float* __restrict__ x, _Float16* __restrict__ h, int n4) {
  int i = blockIdx.x * 256 + threadIdx.x;
  if (i < n4) {
    float4 v = ((const float4*)x)[i];
    v4h o;
    o[0] = (_Float16)v.x; o[1] = (_Float16)v.y; o[2] = (_Float16)v.z; o[3] = (_Float16)v.w;
    ((v4h*)h)[i] = o;
  }
}

// build transposed concat weights: wt[n*256 + k] = (k<128 ? Wrel[k][n] : Wroot[k-128][n]) as f16
__global__ __launch_bounds__(256) void k_cvt_w(const float* __restrict__ wrel,
                                               const float* __restrict__ wroot,
                                               _Float16* __restrict__ wt) {
  int idx = blockIdx.x * 256 + threadIdx.x;   // 128*256
  if (idx >= DIMH * KTOT) return;
  int n = idx & (DIMH - 1);
  int k = idx >> 7;
  float w = (k < DIMH) ? wrel[k * DIMH + n] : wroot[(k - DIMH) * DIMH + n];
  wt[n * KTOT + k] = (_Float16)w;
}

// ---------------- pull-style neighbor aggregation (no float atomics) ----------------
// one wave per node; lane handles 4 columns (8B loads), f32 accumulation, f16 store
__global__ __launch_bounds__(256) void k_gather(const int* __restrict__ rowptr,
                                                const int* __restrict__ eidx,
                                                const _Float16* __restrict__ hin,
                                                _Float16* __restrict__ agg) {
  int node = (blockIdx.x * 256 + threadIdx.x) >> 5;
  int lane = threadIdx.x & 31;
  if (node >= NNODES) return;
  int beg = rowptr[node], end = rowptr[node + 1];
  int c = lane * 4;
  float a0 = 0.f, a1 = 0.f, a2 = 0.f, a3 = 0.f;
  for (int e = beg; e < end; ++e) {
    int s = eidx[e];
    v4h hv = *(const v4h*)(hin + (size_t)s * DIMH + c);
    a0 += (float)hv[0]; a1 += (float)hv[1]; a2 += (float)hv[2]; a3 += (float)hv[3];
  }
  v4h o;
  o[0] = (_Float16)a0; o[1] = (_Float16)a1; o[2] = (_Float16)a2; o[3] = (_Float16)a3;
  *(v4h*)(agg + (size_t)node * DIMH + c) = o;
}

// ---------------- fused conv GEMM: hout = relu([agg|hin] @ [Wrel;Wroot] + b) ----------------
// block = 256 threads = 8 waves; block computes 16 rows x 128 cols; wave -> one 16x16 WMMA tile
__global__ __launch_bounds__(256) void k_conv(const _Float16* __restrict__ agg,
                                              const _Float16* __restrict__ hin,
                                              const _Float16* __restrict__ wt,   // [128][256] f16, col-major-by-n
                                              const float* __restrict__ bias,
                                              _Float16* __restrict__ hout) {
  __shared__ _Float16 Wlds[DIMH * WSTRIDE];
  __shared__ float blds[DIMH];
  int tid = threadIdx.x;

  // stage weights into LDS with padded stride (b128 loads + b128 ds stores)
  for (int idx = tid; idx < DIMH * (KTOT / 8); idx += 256) {
    int n = idx >> 5;            // 0..127
    int kg = idx & 31;           // 8-half group
    *(v8h*)&Wlds[n * WSTRIDE + kg * 8] = *(const v8h*)(wt + n * KTOT + kg * 8);
  }
  if (tid < DIMH) blds[tid] = bias[tid];
  __syncthreads();

  int m0 = blockIdx.x * 16;          // NNODES % 16 == 0 -> no bounds checks
  int wave = tid >> 5, lane = tid & 31;
  int n0 = wave * 16;
  int mm = lane & 15, hh = lane >> 4;

  v8f c = {};
  const _Float16* arow = agg + (size_t)(m0 + mm) * DIMH;
  const _Float16* hrow = hin + (size_t)(m0 + mm) * DIMH;
  const _Float16* bcol = &Wlds[(n0 + mm) * WSTRIDE];

#pragma unroll
  for (int kt = 0; kt < 8; ++kt) {
    int kb = kt * 32;
    // A fragment: lane(m,h) holds K in [kb+8h, kb+8h+8) and [kb+16+8h, kb+16+8h+8)
    const _Float16* s = (kt < 4) ? (arow + kb) : (hrow + (kb - DIMH));
    v8h alo = *(const v8h*)(s + 8 * hh);
    v8h ahi = *(const v8h*)(s + 16 + 8 * hh);
    v16h a = __builtin_shufflevector(alo, ahi, 0, 1, 2, 3, 4, 5, 6, 7, 8, 9, 10, 11, 12, 13, 14, 15);
    // B fragment: lane(n,h) holds column n0+n, K in [kb+8h, kb+8h+8) and [kb+16+8h, kb+16+8h+8)
    const _Float16* bp = bcol + kb + 8 * hh;
    v8h blo = *(const v8h*)bp;
    v8h bhi = *(const v8h*)(bp + 16);
    v16h b = __builtin_shufflevector(blo, bhi, 0, 1, 2, 3, 4, 5, 6, 7, 8, 9, 10, 11, 12, 13, 14, 15);
    c = __builtin_amdgcn_wmma_f32_16x16x32_f16(false, a, false, b, (short)0, c, false, false);
  }

  // epilogue: C element (v, lane) -> M = v + 8*hh, N = n0 + (lane&15)
  int n = n0 + mm;
  float bn = blds[n];
  _Float16* orow = hout + (size_t)(m0 + 8 * hh) * DIMH + n;
#pragma unroll
  for (int v = 0; v < 8; ++v) {
    float r = c[v] + bn;
    r = r > 0.0f ? r : 0.0f;
    orow[(size_t)v * DIMH] = (_Float16)r;
  }
}

// ---------------- pooling + FC head ----------------
__global__ __launch_bounds__(256) void k_pool(const int* __restrict__ batch,
                                              const _Float16* __restrict__ h,
                                              float* __restrict__ sums, float* __restrict__ cnt) {
  int node = (blockIdx.x * 256 + threadIdx.x) >> 5;
  int lane = threadIdx.x & 31;
  if (node >= NNODES) return;
  int g = batch[node];
  int c = lane * 4;
  v4h hv = *(const v4h*)(h + (size_t)node * DIMH + c);
  float* s = sums + (size_t)g * DIMH + c;
  atomicAdd(&s[0], (float)hv[0]);
  atomicAdd(&s[1], (float)hv[1]);
  atomicAdd(&s[2], (float)hv[2]);
  atomicAdd(&s[3], (float)hv[3]);
  if (lane == 0) atomicAdd(&cnt[g], 1.0f);
}

__global__ __launch_bounds__(256) void k_fc(const float* __restrict__ sums,
                                            const float* __restrict__ cnt,
                                            const float* __restrict__ fcw,
                                            const float* __restrict__ fcb,
                                            float* __restrict__ out1) {
  int i = blockIdx.x * 256 + threadIdx.x;
  if (i >= NGRAPHS * DIMFC) return;
  int g = i / DIMFC, j = i % DIMFC;
  float cg = cnt[g];
  float inv = 1.0f / (cg > 1.0f ? cg : 1.0f);
  float acc = fcb[j];
  const float* sr = sums + (size_t)g * DIMH;
  for (int k = 0; k < DIMH; ++k) acc += (sr[k] * inv) * fcw[k * DIMFC + j];
  out1[i] = acc;
}

__global__ __launch_bounds__(256) void k_reg(const float* __restrict__ out1,
                                             const float* __restrict__ regw,
                                             const float* __restrict__ regb,
                                             float* __restrict__ out) {
  int i = blockIdx.x * 256 + threadIdx.x;
  if (i >= NGRAPHS * DIMOUT) return;
  int g = i / DIMOUT, o = i % DIMOUT;
  float acc = regb[o];
  const float* r = out1 + (size_t)g * DIMFC;
  for (int k = 0; k < DIMFC; ++k) acc += r[k] * regw[k * DIMOUT + o];
  out[i] = acc;
}

// ---------------- driver ----------------
extern "C" void kernel_launch(void* const* d_in, const int* in_sizes, int n_in,
                              void* d_out, int out_size, void* d_ws, size_t ws_size,
                              hipStream_t stream) {
  (void)in_sizes; (void)n_in; (void)out_size; (void)ws_size;
  const float* x     = (const float*)d_in[0];
  const int*   ei    = (const int*)d_in[1];
  const int*   src   = ei;
  const int*   dst   = ei + NEDGES;
  const int*   batch = (const int*)d_in[2];
  const float* Wrel[4];  const float* Wroot[4];  const float* bv[4];
  for (int l = 0; l < 4; ++l) {
    Wrel[l]  = (const float*)d_in[3 + 3 * l];
    Wroot[l] = (const float*)d_in[4 + 3 * l];
    bv[l]    = (const float*)d_in[5 + 3 * l];
  }
  const float* fcw  = (const float*)d_in[15];
  const float* fcb  = (const float*)d_in[16];
  const float* regw = (const float*)d_in[17];
  const float* regb = (const float*)d_in[18];
  float* out = (float*)d_out;

  char* ws = (char*)d_ws;
  size_t off = 0;
  auto alloc = [&](size_t bytes) -> char* {
    char* p = ws + off;
    off += (bytes + 255) & ~(size_t)255;
    return p;
  };
  int*      rowptr = (int*)alloc((NNODES + 1) * sizeof(int));
  int*      cursor = (int*)alloc((size_t)NNODES * sizeof(int));     // also histogram counts
  int*      eidx   = (int*)alloc((size_t)NEDGES * sizeof(int));
  int*      bsums  = (int*)alloc(256 * sizeof(int));
  _Float16* hA     = (_Float16*)alloc((size_t)NNODES * DIMH * 2);
  _Float16* hB     = (_Float16*)alloc((size_t)NNODES * DIMH * 2);
  _Float16* agg    = (_Float16*)alloc((size_t)NNODES * DIMH * 2);
  _Float16* wt     = (_Float16*)alloc((size_t)DIMH * KTOT * 2);
  float*    sums   = (float*)alloc((size_t)NGRAPHS * DIMH * 4);
  float*    cnt    = (float*)alloc((size_t)NGRAPHS * 4);
  float*    fc1    = (float*)alloc((size_t)NGRAPHS * DIMFC * 4);

  // ---- CSR build (deterministic set membership; fill order immaterial for sums) ----
  k_zero_i<<<(NNODES + 255) / 256, 256, 0, stream>>>(cursor, NNODES);
  k_hist<<<(NEDGES + 255) / 256, 256, 0, stream>>>(dst, cursor);
  int nScanBlocks = (NNODES + 511) / 512;
  k_scan1<<<nScanBlocks, 512, 0, stream>>>(cursor, rowptr, bsums, NNODES);
  k_scan2<<<1, 32, 0, stream>>>(bsums, nScanBlocks);
  k_scan3<<<(NNODES + 1 + 255) / 256, 256, 0, stream>>>(rowptr, cursor, bsums, NNODES, NEDGES);
  k_fill<<<(NEDGES + 255) / 256, 256, 0, stream>>>(src, dst, cursor, eidx);

  // ---- activations to f16 ----
  k_cvt_x<<<(NNODES * DIMH / 4 + 255) / 256, 256, 0, stream>>>(x, hA, NNODES * DIMH / 4);

  // ---- 4 GraphConv layers (ping-pong hA/hB) ----
  _Float16* hin = hA;
  _Float16* hout = hB;
  for (int l = 0; l < 4; ++l) {
    k_cvt_w<<<(DIMH * KTOT + 255) / 256, 256, 0, stream>>>(Wrel[l], Wroot[l], wt);
    k_gather<<<(NNODES * 32 + 255) / 256, 256, 0, stream>>>(rowptr, eidx, hin, agg);
    k_conv<<<NNODES / 16, 256, 0, stream>>>(agg, hin, wt, bv[l], hout);
    _Float16* t = hin; hin = hout; hout = t;
  }

  // ---- mean pool + FC head ----
  k_zero_f<<<(NGRAPHS * DIMH + 255) / 256, 256, 0, stream>>>(sums, NGRAPHS * DIMH);
  k_zero_f<<<(NGRAPHS + 255) / 256, 256, 0, stream>>>(cnt, NGRAPHS);
  k_pool<<<(NNODES * 32 + 255) / 256, 256, 0, stream>>>(batch, hin, sums, cnt);
  k_fc<<<(NGRAPHS * DIMFC + 255) / 256, 256, 0, stream>>>(sums, cnt, fcw, fcb, fc1);
  k_reg<<<(NGRAPHS * DIMOUT + 255) / 256, 256, 0, stream>>>(fc1, regw, regb, out);
}
